// AttentionPropagation_2559800509228
// MI455X (gfx1250) — matrix operand order repaired and checked by reading, not compiled, not run on hardware
//
#include <hip/hip_runtime.h>
#include <hip/hip_bf16.h>

// ---------------------------------------------------------------------------
// AttentionPropagation fused pipeline for gfx1250 (MI455X), wave32 + WMMA.
//   B=4, C=256, H=4, D=64, N=M=2048
// All GEMMs via V_WMMA_F32_16X16X32_F16 (f16 in, f32 accumulate).
// Attention is flash-style (no [B,H,N,M] materialization).
// GEMM uses 2x2 register blocking per wave: 4 WMMAs per 4 fragment loads.
// ---------------------------------------------------------------------------

typedef __attribute__((ext_vector_type(16))) _Float16 v16h;
typedef __attribute__((ext_vector_type(8)))  _Float16 v8h;
typedef __attribute__((ext_vector_type(8)))  float    v8f;

#define BB 4
#define CC 256
#define HH 4
#define DD 64
#define NN 2048
#define MM 2048

// ---- WMMA fragment loaders (CDNA5 ISA 7.12.2 layouts, wave32) -------------
// A (16x32 f16): lane L -> row (L&15); halves 0..7 = K[8*(L>>4) .. +7],
//                halves 8..15 = K[16+8*(L>>4) .. +7].  Row-major source, ld halves.
__device__ __forceinline__ v16h load_a16(const _Float16* __restrict__ p, int ld) {
  int L = threadIdx.x & 31;
  const _Float16* q = p + (size_t)(L & 15) * ld + 8 * (L >> 4);
  union { v16h v; v8h h[2]; } r;
  r.h[0] = *(const v8h*)q;
  r.h[1] = *(const v8h*)(q + 16);
  return r.v;
}

// B (32x16 f16): lane L -> col (L&15); halves 0..7 = K[8*(L>>4) .. +7],
//                halves 8..15 = K[16+8*(L>>4) .. +7].
// Source layout: element (k, col) at p[col*ld + k]  (K contiguous).
__device__ __forceinline__ v16h load_b16(const _Float16* __restrict__ p, size_t ld) {
  int L = threadIdx.x & 31;
  const _Float16* q = p + (size_t)(L & 15) * ld + 8 * (L >> 4);
  union { v16h v; v8h h[2]; } r;
  r.h[0] = *(const v8h*)q;
  r.h[1] = *(const v8h*)(q + 16);
  return r.v;
}

__device__ __forceinline__ v8f wmma_f16(v16h a, v16h b, v8f c) {
  return __builtin_amdgcn_wmma_f32_16x16x32_f16(
      /*neg_a=*/false, a, /*neg_b=*/false, b,
      /*c_mod=*/(short)0, c, /*reuse_a=*/false, /*reuse_b=*/false);
}

// ---- small helpers --------------------------------------------------------
__global__ void f32_to_f16(const float* __restrict__ in, _Float16* __restrict__ out, int n) {
  int i = blockIdx.x * blockDim.x + threadIdx.x;
  if (i < n) out[i] = (_Float16)in[i];
}

__global__ void bn_prep(const float* __restrict__ gamma, const float* __restrict__ beta,
                        const float* __restrict__ mean, const float* __restrict__ var,
                        float* __restrict__ scale, float* __restrict__ shift, int n) {
  int i = blockIdx.x * blockDim.x + threadIdx.x;
  if (i < n) {
    float s = gamma[i] * rsqrtf(var[i] + 1e-5f);
    scale[i] = s;
    shift[i] = beta[i] - mean[i] * s;
  }
}

// x [b][c][n] f32  ->  out[b][n][ldo] f16 at column offset coloff (c contiguous)
__global__ void transpose_cn_to_nc(const float* __restrict__ x, _Float16* __restrict__ out,
                                   int C, int Nn, int ldo, int coloff) {
  __shared__ _Float16 tile[32][33];
  int b = blockIdx.z;
  int c0 = blockIdx.y * 32, n0 = blockIdx.x * 32;
  const float* xb = x + (size_t)b * C * Nn;
  for (int i = threadIdx.y; i < 32; i += 8)
    tile[i][threadIdx.x] = (_Float16)xb[(size_t)(c0 + i) * Nn + n0 + threadIdx.x];
  __syncthreads();
  _Float16* ob = out + (size_t)b * Nn * ldo + coloff;
  for (int i = threadIdx.y; i < 32; i += 8)
    ob[(size_t)(n0 + i) * ldo + c0 + threadIdx.x] = tile[threadIdx.x][i];
}

// ---- generic WMMA GEMM: Y[b][o][n] = sum_c W[o][c] * X[b][n][c]  ----------
// One 32x32 output block per wave (2x2 register tiles): per K-step, 2 A-frags
// + 2 B-frags feed 4 WMMAs (2x fragment reuse vs one tile per wave).
// Epilogue modes:
//   0: f16 store [b][n][ldo] at +coloff           (bias)
//   1: like 0 with BN scale/shift + ReLU          (bias, bnscale, bnshift)
//   2: f32 store [b][o][n] + residual             (bias, resid -> out32)
//   3: f16 store [b][o][n]                        (bias)
__global__ void __launch_bounds__(256) gemm_wmma(
    const _Float16* __restrict__ W, const float* __restrict__ bias,
    const _Float16* __restrict__ X, int ldx, int CIN, int Npts, int O,
    int mode,
    _Float16* __restrict__ out16, int ldo, int coloff,
    float* __restrict__ out32, const float* __restrict__ resid,
    const float* __restrict__ bnscale, const float* __restrict__ bnshift)
{
  const int wave = threadIdx.x >> 5;
  const int L    = threadIdx.x & 31;
  const int nBo = O >> 5, nBn = Npts >> 5;          // 32-wide blocks
  int t = blockIdx.x * 8 + wave;
  if (t >= BB * nBo * nBn) return;                  // wave-uniform: EXEC stays full
  int b  = t / (nBo * nBn);
  int rr = t % (nBo * nBn);
  int o0 = (rr / nBn) << 5;
  int n0 = (rr % nBn) << 5;

  const _Float16* Xb = X + (size_t)b * Npts * ldx + (size_t)n0 * ldx;
  const _Float16* Wb = W + (size_t)o0 * CIN;

  v8f acc[2][2] = {{{}, {}}, {{}, {}}};
  for (int c0 = 0; c0 < CIN; c0 += 32) {
    v16h a0 = load_a16(Wb + c0, CIN);                       // rows o0..o0+15
    v16h a1 = load_a16(Wb + (size_t)16 * CIN + c0, CIN);    // rows o0+16..o0+31
    v16h x0 = load_b16(Xb + c0, (size_t)ldx);               // cols n0..n0+15
    v16h x1 = load_b16(Xb + (size_t)16 * ldx + c0, (size_t)ldx);
    acc[0][0] = wmma_f16(a0, x0, acc[0][0]);
    acc[0][1] = wmma_f16(a0, x1, acc[0][1]);
    acc[1][0] = wmma_f16(a1, x0, acc[1][0]);
    acc[1][1] = wmma_f16(a1, x1, acc[1][1]);
  }

  const int colL = L & 15, halfL = L >> 4;
#pragma unroll
  for (int i = 0; i < 2; ++i) {
#pragma unroll
    for (int j = 0; j < 2; ++j) {
      const int n  = n0 + 16 * j + colL;       // C/D: lane -> column n
      const int ob = o0 + 16 * i + 8 * halfL;  // C/D: vgpr r -> row ob + r
      if (mode == 0 || mode == 1) {
        _Float16* dst = out16 + (size_t)b * Npts * ldo + (size_t)n * ldo + coloff + ob;
#pragma unroll
        for (int r = 0; r < 8; ++r) {
          float v = acc[i][j][r] + bias[ob + r];
          if (mode == 1) {
            v = v * bnscale[ob + r] + bnshift[ob + r];
            v = v > 0.f ? v : 0.f;
          }
          dst[r] = (_Float16)v;
        }
      } else if (mode == 3) {
#pragma unroll
        for (int r = 0; r < 8; ++r) {
          float v = acc[i][j][r] + bias[ob + r];
          out16[(size_t)b * O * Npts + (size_t)(ob + r) * Npts + n] = (_Float16)v;
        }
      } else { // mode 2: f32 [b][o][n] + residual
#pragma unroll
        for (int r = 0; r < 8; ++r) {
          size_t idx = (size_t)b * O * Npts + (size_t)(ob + r) * Npts + n;
          out32[idx] = acc[i][j][r] + bias[ob + r] + resid[idx];
        }
      }
    }
  }
}

// ---- flash attention ------------------------------------------------------
// Block: 256 threads (8 waves) handles 64 query rows of one (b, h).
// Wave w: nsub = w&3 (16-row strip), msel = w>>2 (which 16-col half of the
// 32-wide M chunk for S; which pair of 16-col d tiles for O).
__global__ void __launch_bounds__(256) flash_attn(
    const _Float16* __restrict__ qT,   // [B][N][C], c contiguous
    const _Float16* __restrict__ kT,   // [B][M][C], c contiguous
    const _Float16* __restrict__ vT,   // [B][C][M], m contiguous
    const unsigned char* __restrict__ mask, // [B][M] bool
    _Float16* __restrict__ attnT)      // [B][N][C]
{
  __shared__ float Sbuf[64][33];
  __shared__ __attribute__((aligned(16))) _Float16 Pbuf[64][32];
  __shared__ float corrLds[64];
  __shared__ float linvLds[64];

  const int n0   = blockIdx.x * 64;
  const int h    = blockIdx.y;
  const int b    = blockIdx.z;
  const int tid  = threadIdx.x;
  const int wave = tid >> 5;
  const int L    = tid & 31;
  const int colL = L & 15, halfL = L >> 4;
  const int nsub = wave & 3;
  const int msel = wave >> 2;

  // Q fragments for this wave's 16 rows: resident for the whole M loop.
  const _Float16* qbase = qT + ((size_t)b * NN + n0 + 16 * nsub) * CC + h * DD;
  const v16h qf0 = load_a16(qbase, CC);        // K = d 0..31
  const v16h qf1 = load_a16(qbase + 32, CC);   // K = d 32..63

  v8f oa = {}, ob = {};                        // O tiles at d0 = 16*msel, 16*(msel+2)
  float mrun = -INFINITY, lrun = 0.f;          // valid in threads tid<64 (row = tid)

  for (int m0 = 0; m0 < MM; m0 += 32) {
    // ---- S tile (16 rows x 16 cols): S = Q . K^T over d=64 (2 WMMAs)
    const _Float16* kb = kT + ((size_t)b * MM + m0 + 16 * msel) * CC + h * DD;
    v8f s = {};
    s = wmma_f16(qf0, load_b16(kb, CC), s);
    s = wmma_f16(qf1, load_b16(kb + 32, CC), s);

    // mask (applied pre-scale as -1e6, then /sqrt(D)=8 -> -1.25e5), scale
    const int  mcol = m0 + 16 * msel + colL;
    const bool keep = mask[(size_t)b * MM + mcol] != 0;
    const int  srow = 16 * nsub + 8 * halfL;
    const int  scol = 16 * msel + colL;
#pragma unroll
    for (int r = 0; r < 8; ++r)
      Sbuf[srow + r][scol] = keep ? s[r] * 0.125f : -1.25e5f;
    __syncthreads();

    // ---- online softmax: thread tid<64 owns row tid of the 64x32 chunk
    if (tid < 64) {
      float cmax = -INFINITY;
#pragma unroll
      for (int j = 0; j < 32; ++j) cmax = fmaxf(cmax, Sbuf[tid][j]);
      float mnew = fmaxf(mrun, cmax);
      float corr = __expf(mrun - mnew);        // first iter: exp(-inf)=0
      float ssum = 0.f;
#pragma unroll
      for (int j = 0; j < 32; ++j) {
        float p = __expf(Sbuf[tid][j] - mnew);
        ssum += p;
        Pbuf[tid][j] = (_Float16)p;
      }
      lrun = lrun * corr + ssum;
      mrun = mnew;
      corrLds[tid] = corr;
    }
    __syncthreads();

    // ---- O update: rescale by corr, then O += P (A, 16x32) . V^T (B, 32x16)
    float cr[8];
#pragma unroll
    for (int r = 0; r < 8; ++r) cr[r] = corrLds[16 * nsub + r + 8 * halfL];
#pragma unroll
    for (int r = 0; r < 8; ++r) { oa[r] *= cr[r]; ob[r] *= cr[r]; }

    v16h pa = load_a16(&Pbuf[16 * nsub][0], 32);
    const _Float16* vb0 = vT + ((size_t)b * CC + h * DD + 16 * msel)       * MM + m0;
    const _Float16* vb1 = vT + ((size_t)b * CC + h * DD + 16 * (msel + 2)) * MM + m0;
    oa = wmma_f16(pa, load_b16(vb0, MM), oa);
    ob = wmma_f16(pa, load_b16(vb1, MM), ob);
    __syncthreads();   // Pbuf/corr consumed before next chunk overwrites
  }

  if (tid < 64) linvLds[tid] = 1.f / lrun;
  __syncthreads();

#pragma unroll
  for (int r = 0; r < 8; ++r) {
    int   nrow = n0 + 16 * nsub + r + 8 * halfL;
    float li   = linvLds[16 * nsub + r + 8 * halfL];
    size_t base = ((size_t)b * NN + nrow) * CC + h * DD;
    attnT[base + 16 * msel + colL]       = (_Float16)(oa[r] * li);
    attnT[base + 16 * (msel + 2) + colL] = (_Float16)(ob[r] * li);
  }
}

// ---------------------------------------------------------------------------
extern "C" void kernel_launch(void* const* d_in, const int* in_sizes, int n_in,
                              void* d_out, int out_size, void* d_ws, size_t ws_size,
                              hipStream_t stream) {
  (void)in_sizes; (void)n_in; (void)out_size; (void)ws_size;

  const float* x1 = (const float*)d_in[0];
  const float* x2 = (const float*)d_in[1];
  const unsigned char* kv_mask = (const unsigned char*)d_in[2];
  const float* Wq  = (const float*)d_in[3];  const float* bq  = (const float*)d_in[4];
  const float* Wk  = (const float*)d_in[5];  const float* bk  = (const float*)d_in[6];
  const float* Wv  = (const float*)d_in[7];  const float* bv  = (const float*)d_in[8];
  const float* Wmh = (const float*)d_in[9];  const float* bmh = (const float*)d_in[10];
  const float* W1  = (const float*)d_in[11]; const float* b1  = (const float*)d_in[12];
  const float* bn_gamma = (const float*)d_in[13];
  const float* bn_beta  = (const float*)d_in[14];
  const float* bn_mean  = (const float*)d_in[15];
  const float* bn_var   = (const float*)d_in[16];
  const float* W2  = (const float*)d_in[17]; const float* b2  = (const float*)d_in[18];
  float* out = (float*)d_out;

  // ---- workspace carve-up (all offsets deterministic)
  char* wsp = (char*)d_ws;
  auto alloc = [&](size_t bytes) -> void* {
    void* p = (void*)wsp;
    wsp += (bytes + 255) & ~(size_t)255;
    return p;
  };
  _Float16* Wq16  = (_Float16*)alloc((size_t)CC * CC * 2);
  _Float16* Wk16  = (_Float16*)alloc((size_t)CC * CC * 2);
  _Float16* Wv16  = (_Float16*)alloc((size_t)CC * CC * 2);
  _Float16* Wmh16 = (_Float16*)alloc((size_t)CC * CC * 2);
  _Float16* W116  = (_Float16*)alloc((size_t)2 * CC * 2 * CC * 2);
  _Float16* W216  = (_Float16*)alloc((size_t)CC * 2 * CC * 2);
  float*    bnS   = (float*)alloc((size_t)2 * CC * 4);
  float*    bnT   = (float*)alloc((size_t)2 * CC * 4);
  _Float16* catT  = (_Float16*)alloc((size_t)BB * NN * 2 * CC * 2); // [b][n][512]: x1 | mh
  _Float16* x2T   = (_Float16*)alloc((size_t)BB * MM * CC * 2);     // [b][m][256]
  _Float16* qT    = (_Float16*)alloc((size_t)BB * NN * CC * 2);     // [b][n][256]
  _Float16* kT    = (_Float16*)alloc((size_t)BB * MM * CC * 2);     // [b][m][256]
  _Float16* vT    = (_Float16*)alloc((size_t)BB * CC * MM * 2);     // [b][c][m]
  _Float16* attnT = (_Float16*)alloc((size_t)BB * NN * CC * 2);     // [b][n][256]
  _Float16* hT    = (_Float16*)alloc((size_t)BB * NN * 2 * CC * 2); // [b][n][512]

  // ---- weight / BN prep
  f32_to_f16<<<CC * CC / 256, 256, 0, stream>>>(Wq,  Wq16,  CC * CC);
  f32_to_f16<<<CC * CC / 256, 256, 0, stream>>>(Wk,  Wk16,  CC * CC);
  f32_to_f16<<<CC * CC / 256, 256, 0, stream>>>(Wv,  Wv16,  CC * CC);
  f32_to_f16<<<CC * CC / 256, 256, 0, stream>>>(Wmh, Wmh16, CC * CC);
  f32_to_f16<<<4 * CC * CC / 256, 256, 0, stream>>>(W1, W116, 4 * CC * CC);
  f32_to_f16<<<2 * CC * CC / 256, 256, 0, stream>>>(W2, W216, 2 * CC * CC);
  bn_prep<<<2, 256, 0, stream>>>(bn_gamma, bn_beta, bn_mean, bn_var, bnS, bnT, 2 * CC);

  // ---- transposes: x1 -> catT[:, :, 0:256], x2 -> x2T
  dim3 tb(32, 8);
  transpose_cn_to_nc<<<dim3(NN / 32, CC / 32, BB), tb, 0, stream>>>(x1, catT, CC, NN, 2 * CC, 0);
  transpose_cn_to_nc<<<dim3(MM / 32, CC / 32, BB), tb, 0, stream>>>(x2, x2T,  CC, MM, CC, 0);

  // ---- Q/K/V projections (WMMA), 32x32 block per wave
  int jobsP = BB * (CC / 32) * (NN / 32);            // 2048 wave-jobs -> 256 blocks
  gemm_wmma<<<jobsP / 8, 256, 0, stream>>>(Wq16, bq, catT, 2 * CC, CC, NN, CC,
                                           0, qT, CC, 0, nullptr, nullptr, nullptr, nullptr);
  gemm_wmma<<<jobsP / 8, 256, 0, stream>>>(Wk16, bk, x2T, CC, CC, MM, CC,
                                           0, kT, CC, 0, nullptr, nullptr, nullptr, nullptr);
  gemm_wmma<<<jobsP / 8, 256, 0, stream>>>(Wv16, bv, x2T, CC, CC, MM, CC,
                                           3, vT, 0, 0, nullptr, nullptr, nullptr, nullptr);

  // ---- flash attention -> attnT
  flash_attn<<<dim3(NN / 64, HH, BB), 256, 0, stream>>>(qT, kT, vT, kv_mask, attnT);

  // ---- Wmh projection, written into catT[:, :, 256:512] (concat is free)
  gemm_wmma<<<jobsP / 8, 256, 0, stream>>>(Wmh16, bmh, attnT, CC, CC, NN, CC,
                                           0, catT, 2 * CC, CC, nullptr, nullptr, nullptr, nullptr);

  // ---- W1 + BN + ReLU -> hT
  int jobsH = BB * (2 * CC / 32) * (NN / 32);        // 4096 wave-jobs -> 512 blocks
  gemm_wmma<<<jobsH / 8, 256, 0, stream>>>(W116, b1, catT, 2 * CC, 2 * CC, NN, 2 * CC,
                                           1, hT, 2 * CC, 0, nullptr, nullptr, bnS, bnT);

  // ---- W2 + residual -> out (f32 [B,C,N])
  gemm_wmma<<<jobsP / 8, 256, 0, stream>>>(W216, b2, hT, 2 * CC, 2 * CC, NN, CC,
                                           2, nullptr, 0, 0, out, x1, nullptr, nullptr);
}
